// MultiQueryAttention_36902359007579
// MI455X (gfx1250) — matrix operand order repaired and checked by reading, not compiled, not run on hardware
//
#include <hip/hip_runtime.h>

// ---------------------------------------------------------------------------
// MQA forward for MI455X (gfx1250): bf16 WMMA path, f32 accumulation.
// Pipeline: cvt(x) ; cvt+transpose(W*) ; GEMM Q/K/V (TDM-staged, double-
// buffered LDS) ; flash-attn ; GEMM O.
// ---------------------------------------------------------------------------

typedef __attribute__((ext_vector_type(16))) __bf16       v16bf;
typedef __attribute__((ext_vector_type(8)))  float        v8f;
typedef __attribute__((ext_vector_type(4)))  unsigned int u32x4;
typedef __attribute__((ext_vector_type(8)))  int          i32x8;
typedef __attribute__((ext_vector_type(4)))  int          i32x4;

union FragU { u32x4 q[2]; v16bf v; };

__device__ __forceinline__ unsigned short f2bf(float f) {
    unsigned u = __float_as_uint(f);
    u += 0x7fffu + ((u >> 16) & 1u);       // round-to-nearest-even
    return (unsigned short)(u >> 16);
}

__device__ __forceinline__ v8f zero8() {
    v8f z = {0.f,0.f,0.f,0.f,0.f,0.f,0.f,0.f};
    return z;
}

__device__ __forceinline__ v8f wmma_bf16(const FragU& a, const FragU& b, v8f c) {
    return __builtin_amdgcn_wmma_f32_16x16x32_bf16(false, a.v, false, b.v,
                                                   (short)0, c, false, false);
}

// Issue a TDM 2D tile load: tile_d0 elements/row (2B each), tile_d1 rows,
// row stride = stride_d0 elements. D# per CDNA5 ISA ch.8.
// 6-arg builtin (clang-23 / therock-10.0 lane): groups 0..3 + extra + cpol.
__device__ __forceinline__ void tdm_load_2d(unsigned lds_addr, const void* gptr,
                                            unsigned tile_d0, unsigned tile_d1,
                                            unsigned stride_d0) {
    unsigned long long ga = (unsigned long long)gptr;
    u32x4 g0;
    g0[0] = 1u;                                     // count=1, user mode
    g0[1] = lds_addr;                               // LDS byte address
    g0[2] = (unsigned)(ga & 0xffffffffu);           // global_addr[31:0]
    g0[3] = (unsigned)((ga >> 32) & 0x01ffffffu)    // global_addr[56:32]
          | (2u << 30);                             // type = 2 ("image")
    i32x8 g1;
    g1[0] = (int)(1u << 16);                        // wg_mask=0, data_size=2B
    g1[1] = (int)((tile_d0 & 0xffffu) << 16);       // tensor_dim0 = tile_d0
    g1[2] = (int)((tile_d0 >> 16) | ((tile_d1 & 0xffffu) << 16)); // tensor_dim1
    g1[3] = (int)((tile_d1 >> 16) | (tile_d0 << 16));             // tile_dim0
    g1[4] = (int)tile_d1;                           // tile_dim1, tile_dim2=0
    g1[5] = (int)stride_d0;                         // tensor_dim0_stride[31:0]
    g1[6] = 0;
    g1[7] = 0;
    i32x4 z4 = {0,0,0,0};                           // 2D: groups 2/3 unused
    i32x8 z8 = {0,0,0,0,0,0,0,0};
    __builtin_amdgcn_tensor_load_to_lds(g0, g1, z4, z4, z8, 0);
}

// ---------------------------------------------------------------- conversions
__global__ void k_cvt_bf16(const float* __restrict__ in,
                           unsigned short* __restrict__ out, int n) {
    int i = (blockIdx.x * blockDim.x + threadIdx.x) * 4;
    if (i >= n) return;
    float4 f = *(const float4*)(in + i);
    unsigned long long p =  (unsigned long long)f2bf(f.x)
                         | ((unsigned long long)f2bf(f.y) << 16)
                         | ((unsigned long long)f2bf(f.z) << 32)
                         | ((unsigned long long)f2bf(f.w) << 48);
    *(unsigned long long*)(out + i) = p;
}

// W[K][N] f32 -> Wt[N][K] bf16 (coalesced writes)
__global__ void k_cvt_t(const float* __restrict__ in,
                        unsigned short* __restrict__ out, int K, int N) {
    long long t = (long long)blockIdx.x * blockDim.x + threadIdx.x;
    if (t >= (long long)K * N) return;
    int k  = (int)(t % K);
    int nn = (int)(t / K);
    out[t] = f2bf(in[(long long)k * N + nn]);
}

// ---------------------------------------------------------------- tiled GEMM
// C[M,N] = A[M,K](bf16) * Bt[N,K](bf16, pre-transposed) + bias
// Tiles staged by the Tensor Data Mover into double-buffered LDS.
// MODE 0: bf16 row-major out   MODE 1: bf16 V-transpose out [B][d][S]
// MODE 2: f32 row-major out
template <int MODE>
__global__ __launch_bounds__(256)
void k_gemm(const unsigned short* __restrict__ A,
            const unsigned short* __restrict__ Bt,
            const float* __restrict__ bias,
            void* __restrict__ outp, int N, int K) {
    __shared__ __align__(16) unsigned short As[2][128 * 32];
    __shared__ __align__(16) unsigned short Bs[2][128 * 32];

    const int tid  = threadIdx.x;
    const int lane = tid & 31;
    const int wid  = tid >> 5;
    const int l16  = lane & 15;
    const int hi   = lane >> 4;                 // 0: lanes 0-15, 1: lanes 16-31
    const int m0   = blockIdx.y * 128;
    const int n0   = blockIdx.x * 128;
    const int wm0  = (wid >> 2) * 64;           // 2x4 wave grid -> 64x32 / wave
    const int wn0  = (wid & 3) * 32;
    const int akb  = hi ? 8 : 0;                // A-frag K base (halves)
    const int bkb  = hi ? 8 : 0;                // B-frag K base (halves)

    const unsigned ldsA = (unsigned)(unsigned long long)(void*)&As[0][0];
    const unsigned ldsB = (unsigned)(unsigned long long)(void*)&Bs[0][0];

    v8f acc[4][2];
#pragma unroll
    for (int i = 0; i < 4; i++)
#pragma unroll
        for (int j = 0; j < 2; j++) acc[i][j] = zero8();

    const int nk = K >> 5;                      // 32-wide K chunks
    if (wid == 0) {                             // prologue: stage chunk 0
        tdm_load_2d(ldsA, A  + (long long)m0 * K, 32, 128, (unsigned)K);
        tdm_load_2d(ldsB, Bt + (long long)n0 * K, 32, 128, (unsigned)K);
    }

    for (int kt = 0; kt < nk; kt++) {
        const int buf = kt & 1;
        if (wid == 0) {
            if (kt + 1 < nk) {                  // stage next chunk, then wait
                const int nb = (kt + 1) & 1;    // for current pair (in-order)
                const long long k1 = (long long)(kt + 1) << 5;
                tdm_load_2d(ldsA + nb * 8192, A  + (long long)m0 * K + k1,
                            32, 128, (unsigned)K);
                tdm_load_2d(ldsB + nb * 8192, Bt + (long long)n0 * K + k1,
                            32, 128, (unsigned)K);
                __builtin_amdgcn_s_wait_tensorcnt(2);
            } else {
                __builtin_amdgcn_s_wait_tensorcnt(0);
            }
        }
        __syncthreads();

        FragU fa[4], fb[2];
#pragma unroll
        for (int i = 0; i < 4; i++) {
            const unsigned short* ar = &As[buf][(wm0 + i * 16 + l16) * 32];
            fa[i].q[0] = *(const u32x4*)(ar + akb);
            fa[i].q[1] = *(const u32x4*)(ar + akb + 16);
        }
#pragma unroll
        for (int j = 0; j < 2; j++) {
            const unsigned short* br = &Bs[buf][(wn0 + j * 16 + l16) * 32];
            fb[j].q[0] = *(const u32x4*)(br + bkb);
            fb[j].q[1] = *(const u32x4*)(br + bkb + 16);
        }
#pragma unroll
        for (int i = 0; i < 4; i++)
#pragma unroll
            for (int j = 0; j < 2; j++)
                acc[i][j] = wmma_bf16(fa[i], fb[j], acc[i][j]);
        __syncthreads();                        // buffer reuse fence
    }

#pragma unroll
    for (int i = 0; i < 4; i++) {
#pragma unroll
        for (int j = 0; j < 2; j++) {
            const int nn = n0 + wn0 + j * 16 + l16;
            const float bval = bias[nn];
#pragma unroll
            for (int r = 0; r < 8; r++) {
                const int mm  = m0 + wm0 + i * 16 + (hi ? r + 8 : r);
                const float v = acc[i][j][r] + bval;
                if (MODE == 0) {
                    ((unsigned short*)outp)[(long long)mm * N + nn] = f2bf(v);
                } else if (MODE == 1) {           // V: [token][d] -> [B][d][S]
                    const int b = mm >> 11, s = mm & 2047;
                    ((unsigned short*)outp)[((long long)(b * 128 + nn) << 11) + s] = f2bf(v);
                } else {
                    ((float*)outp)[(long long)mm * N + nn] = v;
                }
            }
        }
    }
}

// ---------------------------------------------------------------- flash attn
// One wave = 16 queries of one (batch, head); streams all 2048 keys in
// 32-key chunks with online softmax. Q [BS,2048] bf16, K [BS,128] bf16,
// Vt [B][128][2048] bf16, AO [BS,2048] bf16.
__global__ __launch_bounds__(256)
void k_attn(const unsigned short* __restrict__ Q,
            const unsigned short* __restrict__ Kb,
            const unsigned short* __restrict__ Vt,
            unsigned short* __restrict__ AO) {
    __shared__ __align__(16) unsigned short Plds[8 * 16 * 32];

    const int tid  = threadIdx.x;
    const int lane = tid & 31;
    const int wid  = tid >> 5;
    const int l16  = lane & 15;
    const int hi   = lane >> 4;
    const int w    = blockIdx.x * 8 + wid;      // wave-tile id
    const int b    = w >> 11;                   // 16 heads * 128 q-tiles = 2048
    const int rem  = w & 2047;
    const int h    = rem >> 7;
    const int q0   = (rem & 127) << 4;

    const int akb = hi ? 8 : 0;
    const int bkb = hi ? 8 : 0;

    // persistent Q fragments: 16 queries x d=128 (4 K-chunks of 32)
    FragU qa[4];
    {
        const unsigned short* qr =
            Q + (long long)(b * 2048 + q0 + l16) * 2048 + h * 128;
#pragma unroll
        for (int dc = 0; dc < 4; dc++) {
            qa[dc].q[0] = *(const u32x4*)(qr + dc * 32 + akb);
            qa[dc].q[1] = *(const u32x4*)(qr + dc * 32 + akb + 16);
        }
    }

    float mrow[8], lrow[8];
    v8f o[8];
#pragma unroll
    for (int r = 0; r < 8; r++) { mrow[r] = -1e30f; lrow[r] = 0.f; }
#pragma unroll
    for (int t = 0; t < 8; t++) o[t] = zero8();

    const float scale = 0.08838834764831845f;   // 1/sqrt(128)
    unsigned short* pw = Plds + wid * 512;      // per-wave 16x32 P tile

    for (int kc = 0; kc < 64; kc++) {
        v8f s0 = zero8(), s1 = zero8();
        const unsigned short* kr =
            Kb + (long long)(b * 2048 + kc * 32 + l16) * 128;
#pragma unroll
        for (int dc = 0; dc < 4; dc++) {
            FragU k0f, k1f;
            const unsigned short* p0 = kr + dc * 32 + bkb;
            k0f.q[0] = *(const u32x4*)(p0);
            k0f.q[1] = *(const u32x4*)(p0 + 16);
            const unsigned short* p1 = p0 + 16 * 128;   // keys +16
            k1f.q[0] = *(const u32x4*)(p1);
            k1f.q[1] = *(const u32x4*)(p1 + 16);
            s0 = wmma_bf16(qa[dc], k0f, s0);
            s1 = wmma_bf16(qa[dc], k1f, s1);
        }

        // online softmax: row (= fixed q) lives across one 16-lane half
        float al[8];
#pragma unroll
        for (int r = 0; r < 8; r++) {
            float a = s0[r] * scale, c = s1[r] * scale;
            s0[r] = a; s1[r] = c;
            float t = fmaxf(a, c);
            t = fmaxf(t, __shfl_xor(t, 1, 32));
            t = fmaxf(t, __shfl_xor(t, 2, 32));
            t = fmaxf(t, __shfl_xor(t, 4, 32));
            t = fmaxf(t, __shfl_xor(t, 8, 32));
            const float mn = fmaxf(mrow[r], t);
            al[r] = __expf(mrow[r] - mn);
            mrow[r] = mn;
            const float p0 = __expf(s0[r] - mn);
            const float p1 = __expf(s1[r] - mn);
            float rs = p0 + p1;
            rs += __shfl_xor(rs, 1, 32);
            rs += __shfl_xor(rs, 2, 32);
            rs += __shfl_xor(rs, 4, 32);
            rs += __shfl_xor(rs, 8, 32);
            lrow[r] = lrow[r] * al[r] + rs;
            const int qrow = hi ? r + 8 : r;    // C-layout -> P row index
            pw[qrow * 32 + l16]      = f2bf(p0);
            pw[qrow * 32 + l16 + 16] = f2bf(p1);
        }
        asm volatile("s_wait_dscnt 0" ::: "memory");

#pragma unroll
        for (int t = 0; t < 8; t++)
#pragma unroll
            for (int r = 0; r < 8; r++) o[t][r] *= al[r];

        FragU pa;   // P as A-fragment (16q x 32keys) from LDS
        {
            const unsigned short* pr = pw + l16 * 32;
            pa.q[0] = *(const u32x4*)(pr + akb);
            pa.q[1] = *(const u32x4*)(pr + akb + 16);
        }
#pragma unroll
        for (int nj = 0; nj < 8; nj++) {
            FragU fv;   // Vt[b][nj*16 + n][keys] contiguous in K
            const unsigned short* vr =
                Vt + ((long long)(b * 128 + nj * 16 + l16) << 11) + kc * 32 + bkb;
            fv.q[0] = *(const u32x4*)(vr);
            fv.q[1] = *(const u32x4*)(vr + 16);
            o[nj] = wmma_bf16(pa, fv, o[nj]);
        }
    }

#pragma unroll
    for (int r = 0; r < 8; r++) {
        const float inv = 1.0f / lrow[r];
        const int tok = b * 2048 + q0 + (hi ? r + 8 : r);
        unsigned short* orow = AO + (long long)tok * 2048 + h * 128;
#pragma unroll
        for (int nj = 0; nj < 8; nj++)
            orow[nj * 16 + l16] = f2bf(o[nj][r] * inv);
    }
}

// ---------------------------------------------------------------- launch
extern "C" void kernel_launch(void* const* d_in, const int* in_sizes, int n_in,
                              void* d_out, int out_size, void* d_ws, size_t ws_size,
                              hipStream_t stream) {
    (void)in_sizes; (void)n_in; (void)out_size; (void)ws_size;
    const float* x  = (const float*)d_in[0];
    const float* Wq = (const float*)d_in[1];
    const float* bq = (const float*)d_in[2];
    const float* Wk = (const float*)d_in[3];
    const float* bk = (const float*)d_in[4];
    const float* Wv = (const float*)d_in[5];
    const float* bv = (const float*)d_in[6];
    const float* Wo = (const float*)d_in[7];
    const float* bo = (const float*)d_in[8];

    constexpr int D = 2048, DK = 128, M = 4 * 2048;   // BS = 8192

    char* w = (char*)d_ws;
    unsigned short* xbf = (unsigned short*)(w);                       // 32 MB
    unsigned short* WqT = (unsigned short*)(w + 33554432ll);          //  8 MB
    unsigned short* WkT = (unsigned short*)(w + 41943040ll);          // .5 MB
    unsigned short* WvT = (unsigned short*)(w + 42467328ll);          // .5 MB
    unsigned short* WoT = (unsigned short*)(w + 42991616ll);          //  8 MB
    unsigned short* Qb  = (unsigned short*)(w + 51380224ll);          // 32 MB
    unsigned short* Kbf = (unsigned short*)(w + 84934656ll);          //  2 MB
    unsigned short* Vt  = (unsigned short*)(w + 87031808ll);          //  2 MB
    unsigned short* AO  = (unsigned short*)(w + 89128960ll);          // 32 MB

    // 1) convert inputs to bf16 (weights transposed to [N][K])
    k_cvt_bf16<<<M * D / 1024, 256, 0, stream>>>(x, xbf, M * D);
    k_cvt_t<<<D * D  / 256, 256, 0, stream>>>(Wq, WqT, D, D);
    k_cvt_t<<<D * DK / 256, 256, 0, stream>>>(Wk, WkT, D, DK);
    k_cvt_t<<<D * DK / 256, 256, 0, stream>>>(Wv, WvT, D, DK);
    k_cvt_t<<<D * D  / 256, 256, 0, stream>>>(Wo, WoT, D, D);

    // 2) projections (TDM-staged GEMMs)
    k_gemm<0><<<dim3(D / 128,  M / 128), 256, 0, stream>>>(xbf, WqT, bq, Qb,  D,  D);
    k_gemm<0><<<dim3(DK / 128, M / 128), 256, 0, stream>>>(xbf, WkT, bk, Kbf, DK, D);
    k_gemm<1><<<dim3(DK / 128, M / 128), 256, 0, stream>>>(xbf, WvT, bv, Vt,  DK, D);

    // 3) attention: 512 wave-tiles (16 queries each), 8 waves / block
    k_attn<<<M / 16 * 16 / 8 / 16, 256, 0, stream>>>(Qb, Kbf, Vt, AO);

    // 4) output projection (f32 out + bias)
    k_gemm<2><<<dim3(D / 128, M / 128), 256, 0, stream>>>(AO, WoT, bo, d_out, D, D);
}
